// GraphClsFilm_75952201663101
// MI455X (gfx1250) — matrix operand order repaired and checked by reading, not compiled, not run on hardware
//
#include <hip/hip_runtime.h>
#include <cstdint>
#include <cstddef>

// ---------------------------------------------------------------------------
// CDNA5 (gfx1250, wave32) fused FiLM-MLP.
//   - GEMMs on V_WMMA_F32_16X16X32_BF16 (f32 accumulate)
//   - weights cooperatively double-buffered into LDS with
//     global_load_async_to_lds_b128 (ASYNCcnt), shared by all 4 waves/block
//   - h never touches HBM: x(bf16)->h1->h2 live in per-wave LDS tiles
// ---------------------------------------------------------------------------

typedef __attribute__((ext_vector_type(16))) __bf16 v16bf;
typedef __attribute__((ext_vector_type(8)))  __bf16 v8bf;
typedef __attribute__((ext_vector_type(4)))  __bf16 v4bf;
typedef __attribute__((ext_vector_type(8)))  float  v8f;

#define DIM   256    // node_dim == cond_dim == H
#define ODIM  512    // out_dim
#define TILES_PER_BLOCK 4
#define WSLICE (256 * 32)       // one K-slice of a 256x256 bf16 weight: 16 KB

__device__ __forceinline__ v8f vzero8() {
  v8f z;
  #pragma unroll
  for (int i = 0; i < 8; ++i) z[i] = 0.f;
  return z;
}

// A-fragment (16x32 bf16, M = lane&15) from a row-major bf16 row in LDS.
// lanes 0-15 take K = k0 + {0..15}; lanes 16-31 take K = k0 + {16..31}
__device__ __forceinline__ v16bf load_a_frag(const __bf16* hrow, int k0, int hi) {
  v8bf lo = *(const v8bf*)(hrow + k0 + hi * 16);
  v8bf hh = *(const v8bf*)(hrow + k0 + 8 + hi * 16);
  return __builtin_shufflevector(lo, hh, 0,1,2,3,4,5,6,7,8,9,10,11,12,13,14,15);
}

// Cooperative async stage of one K-slice (all 256 rows x 32 cols) of a
// row-major 256x256 bf16 weight matrix into LDS. 16 KB per slice,
// 1024 16-byte chunks, wave w moves chunks [w*256, w*256+256).
__device__ __forceinline__ void issue_slice(const __bf16* W, int ks, __bf16* dst,
                                            int wave, int lane) {
  uint32_t ldsbase = (uint32_t)(uintptr_t)dst;   // low 32 bits = LDS offset
  #pragma unroll
  for (int it = 0; it < 8; ++it) {
    int c = wave * 256 + it * 32 + lane;         // chunk id
    int row = c >> 2, q = c & 3;                 // 4 chunks per 64B row
    uint32_t goff = (uint32_t)(row * 512 + ks * 64 + q * 16);
    uint32_t loff = ldsbase + (uint32_t)c * 16;
    asm volatile("global_load_async_to_lds_b128 %0, %1, %2"
                 :: "v"(loff), "v"(goff), "s"(W)
                 : "memory");
  }
}

// One 16x256 @ 256x256^T GEMM for the whole block: A rows from per-wave LDS
// tile, B double-buffered through shared wst by all 4 waves.
// 16 independent accumulator chains per K-slice.
__device__ __forceinline__ void gemm_block(const __bf16* __restrict__ W,
                                           const __bf16* hrow, int hi, int nlo,
                                           int wave, int lane,
                                           __bf16* wst, v8f (&acc)[16]) {
  #pragma unroll
  for (int t = 0; t < 16; ++t) acc[t] = vzero8();
  issue_slice(W, 0, wst, wave, lane);
  for (int ks = 0; ks < 8; ++ks) {
    asm volatile("s_wait_asynccnt 0x0" ::: "memory");  // my slice-ks ops landed
    __syncthreads();                                   // everyone's landed
    if (ks < 7) issue_slice(W, ks + 1, wst + ((ks + 1) & 1) * WSLICE, wave, lane);
    v16bf a = load_a_frag(hrow, ks * 32, hi);
    const __bf16* wb = wst + (ks & 1) * WSLICE;
    #pragma unroll
    for (int t = 0; t < 16; ++t) {
      v16bf b = *(const v16bf*)(wb + (t * 16 + nlo) * 32 + hi * 16);
      acc[t] = __builtin_amdgcn_wmma_f32_16x16x32_bf16(
          false, a, false, b, (short)0, acc[t], false, false);
    }
  }
}

// ---------------------------------------------------------------------------
// Weight-norm prep: out[row,:] = v[row,:] * g[row] / ||v[row,:]||   (C = 256)
// ---------------------------------------------------------------------------
__global__ void wn_rows_bf16(const float* __restrict__ v, const float* __restrict__ g,
                             __bf16* __restrict__ out) {
  const int row = blockIdx.x, t = threadIdx.x;
  float x  = v[(size_t)row * DIM + t];
  float ss = x * x;
  #pragma unroll
  for (int m = 16; m > 0; m >>= 1) ss += __shfl_xor(ss, m, 32);
  __shared__ float red[8];
  if ((t & 31) == 0) red[t >> 5] = ss;
  __syncthreads();
  float tot = 0.f;
  #pragma unroll
  for (int i = 0; i < 8; ++i) tot += red[i];
  float scale = g[row] / sqrtf(tot);
  out[(size_t)row * DIM + t] = (__bf16)(x * scale);
}

__global__ void wn_rows_f32(const float* __restrict__ v, const float* __restrict__ g,
                            float* __restrict__ out) {
  const int row = blockIdx.x, t = threadIdx.x;
  float x  = v[(size_t)row * DIM + t];
  float ss = x * x;
  #pragma unroll
  for (int m = 16; m > 0; m >>= 1) ss += __shfl_xor(ss, m, 32);
  __shared__ float red[8];
  if ((t & 31) == 0) red[t >> 5] = ss;
  __syncthreads();
  float tot = 0.f;
  #pragma unroll
  for (int i = 0; i < 8; ++i) tot += red[i];
  float scale = g[row] / sqrtf(tot);
  out[(size_t)row * DIM + t] = x * scale;
}

// ---------------------------------------------------------------------------
// FiLM tables: gb = cond @ Wc^T + b_cond ; gamma = gb[:, :256] + 1, beta = rest
// ---------------------------------------------------------------------------
__global__ void film_kernel(const float* __restrict__ cond, const float* __restrict__ Wc,
                            const float* __restrict__ b_cond,
                            float* __restrict__ gamma, float* __restrict__ beta) {
  const int gi = blockIdx.x;
  const int j  = threadIdx.x;
  __shared__ float csh[DIM];
  csh[j] = cond[(size_t)gi * DIM + j];
  __syncthreads();
  const float* w0 = Wc + (size_t)j * DIM;
  const float* w1 = Wc + (size_t)(j + DIM) * DIM;
  float d0 = 0.f, d1 = 0.f;
  for (int k = 0; k < DIM; ++k) {
    float c = csh[k];
    d0 = fmaf(c, w0[k], d0);
    d1 = fmaf(c, w1[k], d1);
  }
  gamma[(size_t)gi * DIM + j] = d0 + b_cond[j] + 1.0f;
  beta [(size_t)gi * DIM + j] = d1 + b_cond[DIM + j];
}

// ---------------------------------------------------------------------------
// Fused main kernel: 4 waves/block, one 16-row tile per wave.
//   h1 = LN(x @ W1^T); h1 = relu(h1*gamma[bid] + beta[bid])
//   h2 = relu(h1 @ W2^T + b2); out = h2 @ W3^T + b3 (two 256-col passes)
// ---------------------------------------------------------------------------
__global__ __launch_bounds__(32 * TILES_PER_BLOCK)
void fused_mlp(const float* __restrict__ x, const int* __restrict__ bids,
               const __bf16* __restrict__ W1, const __bf16* __restrict__ W2,
               const __bf16* __restrict__ W3,
               const float* __restrict__ gamma, const float* __restrict__ beta,
               const float* __restrict__ b2, const float* __restrict__ b3,
               float* __restrict__ out, int N, int ntiles) {
  __shared__ __attribute__((aligned(128))) __bf16 hbuf_all[TILES_PER_BLOCK * 16 * DIM]; // 32 KB
  __shared__ __attribute__((aligned(128))) __bf16 wst[2 * WSLICE];                      // 32 KB

  const int lane = threadIdx.x & 31;
  const int wave = threadIdx.x >> 5;
  const int mytile = blockIdx.x * TILES_PER_BLOCK + wave;
  const bool valid = (mytile < ntiles);
  const int tile = valid ? mytile : (ntiles - 1);   // clamp: never skip barriers
  const int row0 = tile * 16;
  const int nlo  = lane & 15;
  const int hi   = lane >> 4;
  __bf16* hb = hbuf_all + wave * (16 * DIM);

  // ---- phase 0: stage x tile into LDS as bf16 (coalesced f32 loads) -------
  {
    const float* xt = x + (size_t)row0 * DIM;
    const size_t base = (size_t)row0 * DIM, limit = (size_t)N * DIM;
    #pragma unroll
    for (int j = 0; j < 32; ++j) {
      int e = (j * 32 + lane) * 4;
      float4 vv;
      if (base + (size_t)e + 3 < limit) vv = *(const float4*)(xt + e);
      else vv = make_float4(0.f, 0.f, 0.f, 0.f);
      v4bf p;
      p[0] = (__bf16)vv.x; p[1] = (__bf16)vv.y; p[2] = (__bf16)vv.z; p[3] = (__bf16)vv.w;
      *(v4bf*)(hb + e) = p;
    }
  }
  asm volatile("s_wait_dscnt 0" ::: "memory");

  const __bf16* hrow = hb + nlo * DIM;   // this lane's A-matrix row
  v8f acc[16];

  // ======================= stage 1: h1 = x @ W1^T ==========================
  gemm_block(W1, hrow, hi, nlo, wave, lane, wst, acc);

  // ---- layernorm over each 256-wide row (registers + 16-lane butterfly) ---
  float mu[8], rs[8];
  #pragma unroll
  for (int i = 0; i < 8; ++i) {
    float s1 = 0.f, s2 = 0.f;
    #pragma unroll
    for (int t = 0; t < 16; ++t) { float v = acc[t][i]; s1 += v; s2 += v * v; }
    #pragma unroll
    for (int m = 8; m > 0; m >>= 1) {
      s1 += __shfl_xor(s1, m, 32);
      s2 += __shfl_xor(s2, m, 32);
    }
    float mean = s1 * (1.f / 256.f);
    float var  = s2 * (1.f / 256.f) - mean * mean;
    mu[i] = mean;
    rs[i] = rsqrtf(var + 1e-5f);
  }

  // ---- FiLM (+1 folded into gamma) + relu, write bf16 h1 to LDS -----------
  int bid[8];
  #pragma unroll
  for (int i = 0; i < 8; ++i) {
    int r = row0 + i + 8 * hi;
    bid[i] = bids[r < N ? r : (N - 1)];
  }
  #pragma unroll
  for (int t = 0; t < 16; ++t) {
    int col = t * 16 + nlo;
    #pragma unroll
    for (int i = 0; i < 8; ++i) {
      float gm = gamma[(size_t)bid[i] * DIM + col];
      float bt = beta [(size_t)bid[i] * DIM + col];
      float v  = (acc[t][i] - mu[i]) * rs[i];
      v = fmaxf(fmaf(v, gm, bt), 0.f);
      hb[(i + 8 * hi) * DIM + col] = (__bf16)v;
    }
  }
  asm volatile("s_wait_dscnt 0" ::: "memory");

  // ======================= stage 2: h2 = relu(h1 @ W2^T + b2) ==============
  gemm_block(W2, hrow, hi, nlo, wave, lane, wst, acc);
  #pragma unroll
  for (int t = 0; t < 16; ++t) {
    int col = t * 16 + nlo;
    float bb = b2[col];
    #pragma unroll
    for (int i = 0; i < 8; ++i) {
      float v = fmaxf(acc[t][i] + bb, 0.f);
      hb[(i + 8 * hi) * DIM + col] = (__bf16)v;
    }
  }
  asm volatile("s_wait_dscnt 0" ::: "memory");

  // ======================= stage 3: out = h2 @ W3^T + b3 ===================
  for (int p = 0; p < 2; ++p) {
    gemm_block(W3 + (size_t)p * 256 * DIM, hrow, hi, nlo, wave, lane, wst, acc);
    #pragma unroll
    for (int t = 0; t < 16; ++t) {
      int ocol = p * 256 + t * 16 + nlo;
      float bb = b3[ocol];
      #pragma unroll
      for (int i = 0; i < 8; ++i) {
        int r = row0 + i + 8 * hi;
        if (valid && r < N) out[(size_t)r * ODIM + ocol] = acc[t][i] + bb;
      }
    }
  }
}

// ---------------------------------------------------------------------------
extern "C" void kernel_launch(void* const* d_in, const int* in_sizes, int n_in,
                              void* d_out, int out_size, void* d_ws, size_t ws_size,
                              hipStream_t stream) {
  const float* x      = (const float*)d_in[0];
  const float* cond   = (const float*)d_in[1];
  const int*   bids   = (const int*)  d_in[2];
  const float* v_cond = (const float*)d_in[3];
  const float* g_cond = (const float*)d_in[4];
  const float* b_cond = (const float*)d_in[5];
  const float* v1     = (const float*)d_in[6];
  const float* g1     = (const float*)d_in[7];
  const float* v2     = (const float*)d_in[8];
  const float* g2     = (const float*)d_in[9];
  const float* b2     = (const float*)d_in[10];
  const float* v3     = (const float*)d_in[11];
  const float* g3     = (const float*)d_in[12];
  const float* b3     = (const float*)d_in[13];

  const int N = in_sizes[0] / DIM;   // 200000
  const int G = in_sizes[1] / DIM;   // 256

  // workspace carve-out (~1.5 MB total)
  char* wsp = (char*)d_ws;
  __bf16* W1n = (__bf16*)wsp; wsp += (size_t)DIM  * DIM * sizeof(__bf16);
  __bf16* W2n = (__bf16*)wsp; wsp += (size_t)DIM  * DIM * sizeof(__bf16);
  __bf16* W3n = (__bf16*)wsp; wsp += (size_t)ODIM * DIM * sizeof(__bf16);
  float*  Wc  = (float*) wsp; wsp += (size_t)ODIM * DIM * sizeof(float);
  float*  gam = (float*) wsp; wsp += (size_t)G    * DIM * sizeof(float);
  float*  bet = (float*) wsp; wsp += (size_t)G    * DIM * sizeof(float);

  wn_rows_bf16<<<DIM,  DIM, 0, stream>>>(v1, g1, W1n);
  wn_rows_bf16<<<DIM,  DIM, 0, stream>>>(v2, g2, W2n);
  wn_rows_bf16<<<ODIM, DIM, 0, stream>>>(v3, g3, W3n);
  wn_rows_f32 <<<ODIM, DIM, 0, stream>>>(v_cond, g_cond, Wc);
  film_kernel <<<G,    DIM, 0, stream>>>(cond, Wc, b_cond, gam, bet);

  const int ntiles  = (N + 15) / 16;
  const int nblocks = (ntiles + TILES_PER_BLOCK - 1) / TILES_PER_BLOCK;
  fused_mlp<<<nblocks, 32 * TILES_PER_BLOCK, 0, stream>>>(
      x, bids, W1n, W2n, W3n, gam, bet, b2, b3, (float*)d_out, N, ntiles);
}
